// minGRULM_5617817224111
// MI455X (gfx1250) — matrix operand order repaired and checked by reading, not compiled, not run on hardware
//
#include <hip/hip_runtime.h>
#include <math.h>

// ---------------- CDNA5 WMMA types ----------------
typedef __attribute__((ext_vector_type(16))) __bf16 v16bf;
typedef __attribute__((ext_vector_type(8)))  float  v8f;
typedef int v4i __attribute__((vector_size(16)));    // builtin's expected pointee

#if __has_builtin(__builtin_amdgcn_global_load_async_to_lds_b128)
#define HAVE_ASYNC_LDS 1
#else
#define HAVE_ASYNC_LDS 0
#endif

__device__ __forceinline__ void wait_async_lds() {
#if HAVE_ASYNC_LDS
#if __has_builtin(__builtin_amdgcn_s_wait_asynccnt)
    __builtin_amdgcn_s_wait_asynccnt(0);
#else
    asm volatile("s_wait_asynccnt 0" ::: "memory");
#endif
#endif
}

// 16B global -> LDS copy (async direct-to-LDS on gfx1250, sync fallback)
__device__ __forceinline__ void copy16_async(const __bf16* g, __bf16* l) {
#if HAVE_ASYNC_LDS
    __builtin_amdgcn_global_load_async_to_lds_b128(
        (__attribute__((address_space(1))) v4i*)(uintptr_t)g,
        (__attribute__((address_space(3))) v4i*)(unsigned)(uintptr_t)l, 0, 0);
#else
    *(uint4*)l = *(const uint4*)g;
#endif
}

__device__ __forceinline__ void stage_tile(const __bf16* gA0, const __bf16* gA1,
                                           const __bf16* gB0, const __bf16* gB1,
                                           __bf16* lA0, __bf16* lA1,
                                           __bf16* lB0, __bf16* lB1) {
#if HAVE_ASYNC_LDS
    copy16_async(gA0, lA0);
    copy16_async(gA1, lA1);
    copy16_async(gB0, lB0);
    copy16_async(gB1, lB1);
#else
    // load all, then store all: keeps 4 global loads in flight
    uint4 a0 = *(const uint4*)gA0;
    uint4 a1 = *(const uint4*)gA1;
    uint4 b0 = *(const uint4*)gB0;
    uint4 b1 = *(const uint4*)gB1;
    *(uint4*)lA0 = a0;
    *(uint4*)lA1 = a1;
    *(uint4*)lB0 = b0;
    *(uint4*)lB1 = b1;
#endif
}

__device__ __forceinline__ v16bf load_frag(const __bf16* p0, const __bf16* p1) {
    union { uint4 q[2]; v16bf v; } u;
    u.q[0] = *reinterpret_cast<const uint4*>(p0);
    u.q[1] = *reinterpret_cast<const uint4*>(p1);
    return u.v;
}

// ---------------- elementwise helpers ----------------
__device__ __forceinline__ float softplusf(float t) {
    return fmaxf(t, 0.0f) + log1pf(expf(-fabsf(t)));
}
__device__ __forceinline__ float log_gf(float x) {
    return (x >= 0.0f) ? logf(x + 0.5f) : -softplusf(-x);
}

// ---------------- fp32 -> bf16 convert (x4 vectorized) ----------------
__global__ void cvt_kernel(const float* __restrict__ src, __bf16* __restrict__ dst, int n4) {
    int i = blockIdx.x * 256 + threadIdx.x;
    if (i < n4) {
        float4 f = ((const float4*)src)[i];
        union { __bf16 h[4]; uint2 u; } t;
        t.h[0] = (__bf16)f.x; t.h[1] = (__bf16)f.y;
        t.h[2] = (__bf16)f.z; t.h[3] = (__bf16)f.w;
        *(uint2*)(dst + 4 * (size_t)i) = t.u;
    }
}

// ---------------- embedding gather (float4): x[b,s,:] = emb[token] --------
__global__ void embed_kernel(const int* __restrict__ tokens, const float* __restrict__ emb,
                             float* __restrict__ x) {
    unsigned idx = blockIdx.x * 256 + threadIdx.x;   // M*D/4 threads, D=1024
    unsigned d4 = idx & 255u;
    unsigned t = idx >> 8;
    int tok = tokens[t];
    ((float4*)x)[idx] = ((const float4*)(emb + (size_t)tok * 1024))[d4];
}

// ---------------- depthwise conv1d K=3 SAME + bias -> bf16 ----------------
__global__ void dwconv_kernel(const float* __restrict__ x, const float* __restrict__ w,
                              const float* __restrict__ b, __bf16* __restrict__ y) {
    unsigned idx = blockIdx.x * 256 + threadIdx.x;   // B*S*D
    unsigned d = idx & 1023u;
    unsigned s = (idx >> 10) & 2047u;
    unsigned bb = idx >> 21;
    const float* xc = x + ((size_t)bb * 2048) * 1024 + d;
    float acc = b[d];
    float w0 = w[d], w1 = w[1024 + d], w2 = w[2048 + d];
    if (s > 0)    acc += xc[(size_t)(s - 1) * 1024] * w0;
    acc += xc[(size_t)s * 1024] * w1;
    if (s < 2047) acc += xc[(size_t)(s + 1) * 1024] * w2;
    y[idx] = (__bf16)acc;
}

// ---------------- RMSNorm (per 1024-row) -> bf16 ----------------
__global__ void rmsnorm_kernel(const float* __restrict__ x, const float* __restrict__ w,
                               __bf16* __restrict__ out) {
    const int row = blockIdx.x;
    const int tid = threadIdx.x;                     // 256 threads, 4 elems each
    const float* xr = x + (size_t)row * 1024;
    float v[4]; float ss = 0.0f;
    #pragma unroll
    for (int i = 0; i < 4; ++i) { v[i] = xr[tid + i * 256]; ss += v[i] * v[i]; }
    __shared__ float red[256];
    red[tid] = ss; __syncthreads();
    for (int s = 128; s > 0; s >>= 1) { if (tid < s) red[tid] += red[tid + s]; __syncthreads(); }
    float scale = rsqrtf(red[0] * (1.0f / 1024.0f) + 1.1920929e-07f);
    #pragma unroll
    for (int i = 0; i < 4; ++i) {
        int d = tid + i * 256;
        out[(size_t)row * 1024 + d] = (__bf16)(v[i] * scale * w[d]);
    }
}

// ---------------- minGRU log-space scan; x += h (residual, in-place) -------
__global__ void scan_kernel(const float* __restrict__ kbuf, const float* __restrict__ thbuf,
                            const float* __restrict__ hprev, float* __restrict__ x) {
    int tid = blockIdx.x * 256 + threadIdx.x;        // B*H = 2048 threads
    int h = tid & 1023;
    int b = tid >> 10;
    const float* kp = kbuf + (size_t)b * 2048 * 1024 + h;
    const float* tp = thbuf + (size_t)b * 2048 * 1024 + h;
    float* xp = x + (size_t)b * 2048 * 1024 + h;
    float a = 0.0f;                                   // running cumsum(log_coeffs)
    float m = log_gf(hprev[b * 1024 + h]);            // streaming logsumexp max
    float ssum = 1.0f;                                // scaled sum
    #pragma unroll 4
    for (int s = 0; s < 2048; ++s) {
        float kk  = kp[(size_t)s * 1024];
        float pre = tp[(size_t)s * 1024];
        float log_z  = -softplusf(-kk);
        float log_c  = -softplusf(kk);
        float log_th = log_gf(pre);
        a += log_c;
        float v = (log_z + log_th) - a;
        if (v <= m) { ssum += expf(v - m); }
        else        { ssum = ssum * expf(m - v) + 1.0f; m = v; }
        float hv = expf(a + m + logf(ssum));
        xp[(size_t)s * 1024] += hv;
    }
}

// ---------------- bf16 WMMA GEMM:  C[M,N] = A[M,K] * W[K,N] + bias --------
// EPI 0: store f32            EPI 1: f32 residual in-place
// EPI 2: GELU -> bf16         EPI 3: non-temporal f32 store (logits)
#define BM 128
#define BN 128
#define BKt 32
#define LDA 40     // 80B row stride: conflict-free A frag ds_load_b128
#define LDB 136    // 272B row stride: 2-way worst case for B frags

template <int EPI>
__global__ __launch_bounds__(256)
void gemm_bf16(const __bf16* __restrict__ A, const __bf16* __restrict__ B,
               const float* __restrict__ bias, float* __restrict__ outf,
               __bf16* __restrict__ outh, int M, int N, int K) {
    __shared__ alignas(16) __bf16 As[2][BM * LDA];
    __shared__ alignas(16) __bf16 Bs[2][BKt * LDB];
    const int tid  = threadIdx.x;
    const int n0   = blockIdx.x * BN;
    const int m0   = blockIdx.y * BM;
    const int wave = tid >> 5, lane = tid & 31;
    const int wr = wave >> 2, wc = wave & 3;          // 2x4 wave grid, each 64x32
    const int g  = lane >> 4, ln = lane & 15;

    v8f acc[4][2];
    #pragma unroll
    for (int i = 0; i < 4; ++i)
        #pragma unroll
        for (int j = 0; j < 2; ++j)
            #pragma unroll
            for (int e = 0; e < 8; ++e) acc[i][j][e] = 0.0f;

    // staging assignment: 512 x 16B chunks each for A (4/row) and B (16/row)
    const int rA0 = tid >> 2,          cA0 = tid & 3;
    const int rA1 = (tid + 256) >> 2,  cA1 = (tid + 256) & 3;
    const int rB0 = tid >> 4,          cB0 = tid & 15;
    const int rB1 = (tid + 256) >> 4,  cB1 = (tid + 256) & 15;
    const int lA0 = rA0 * LDA + cA0 * 8, lA1 = rA1 * LDA + cA1 * 8;
    const int lB0 = rB0 * LDB + cB0 * 8, lB1 = rB1 * LDB + cB1 * 8;

    // running global pointers (incremented; no in-loop 64-bit muls)
    const __bf16* gA0 = A + (size_t)(m0 + rA0) * K + cA0 * 8;
    const __bf16* gA1 = A + (size_t)(m0 + rA1) * K + cA1 * 8;
    const __bf16* gB0 = B + (size_t)rB0 * N + n0 + cB0 * 8;
    const __bf16* gB1 = B + (size_t)rB1 * N + n0 + cB1 * 8;
    const size_t stB = (size_t)BKt * N;

    // prologue: stage tile 0 into buffer 0
    stage_tile(gA0, gA1, gB0, gB1, &As[0][lA0], &As[0][lA1], &Bs[0][lB0], &Bs[0][lB1]);
    gA0 += BKt; gA1 += BKt; gB0 += stB; gB1 += stB;

    const int KT = K >> 5;
    for (int kt = 0; kt < KT; ++kt) {
        const int p = kt & 1;
        wait_async_lds();       // my stage(kt) copies landed in LDS
        __syncthreads();        // everyone's stage(kt) landed; prior reads of buf p done
        if (kt + 1 < KT) {      // overlap: stage(kt+1) streams while we compute tile kt
            stage_tile(gA0, gA1, gB0, gB1,
                       &As[p ^ 1][lA0], &As[p ^ 1][lA1], &Bs[p ^ 1][lB0], &Bs[p ^ 1][lB1]);
            gA0 += BKt; gA1 += BKt; gB0 += stB; gB1 += stB;
        }

        // hoist all fragments, then back-to-back WMMAs
        v16bf bfrag[2];
        #pragma unroll
        for (int j = 0; j < 2; ++j) {                  // B frag: lane%16 == n col, k in table order
            const __bf16* bp = &Bs[p][(g * 8) * LDB + wc * 32 + j * 16 + ln];
            #pragma unroll
            for (int s = 0; s < 8; ++s) { bfrag[j][s] = bp[s * LDB]; bfrag[j][s + 8] = bp[(s + 16) * LDB]; }
        }
        v16bf afrag[4];
        #pragma unroll
        for (int i = 0; i < 4; ++i) {                  // A frag per ISA 16-bit 16x32 layout
            const __bf16* ap = &As[p][(wr * 64 + i * 16 + ln) * LDA + g * 8];
            afrag[i] = load_frag(ap, ap + 16);
        }
        #pragma unroll
        for (int i = 0; i < 4; ++i)
            #pragma unroll
            for (int j = 0; j < 2; ++j)
                acc[i][j] = __builtin_amdgcn_wmma_f32_16x16x32_bf16(
                    false, afrag[i], false, bfrag[j], (short)0, acc[i][j], false, false);
    }

    // epilogue: C/D layout — lane col = ln, rows r + 8*g per VGPR r
    #pragma unroll
    for (int i = 0; i < 4; ++i) {
        #pragma unroll
        for (int j = 0; j < 2; ++j) {
            const int col = n0 + wc * 32 + j * 16 + ln;
            const float bv = bias[col];
            #pragma unroll
            for (int r = 0; r < 8; ++r) {
                const int row = m0 + wr * 64 + i * 16 + g * 8 + r;
                float v = acc[i][j][r] + bv;
                size_t idx = (size_t)row * N + col;
                if (EPI == 0) {
                    outf[idx] = v;
                } else if (EPI == 1) {
                    outf[idx] = v + outf[idx];
                } else if (EPI == 2) {
                    float gg = 0.5f * v * (1.0f + erff(v * 0.70710678118654752f));
                    outh[idx] = (__bf16)gg;
                } else {
                    __builtin_nontemporal_store(v, &outf[idx]);   // logits: bypass L2 retention
                }
            }
        }
    }
}

// ---------------------------------------------------------------------------
extern "C" void kernel_launch(void* const* d_in, const int* in_sizes, int n_in,
                              void* d_out, int out_size, void* d_ws, size_t ws_size,
                              hipStream_t stream) {
    (void)in_sizes; (void)n_in; (void)out_size; (void)ws_size;
    const int*   tokens    = (const int*)d_in[0];
    const float* h_prev    = (const float*)d_in[1];
    const float* emb       = (const float*)d_in[2];
    const float* conv_dw_w = (const float*)d_in[3];
    const float* conv_dw_b = (const float*)d_in[4];
    const float* conv_pw_w = (const float*)d_in[5];
    const float* conv_pw_b = (const float*)d_in[6];
    const float* norm1_w   = (const float*)d_in[7];
    const float* gru_z_w   = (const float*)d_in[8];
    const float* gru_z_b   = (const float*)d_in[9];
    const float* gru_h_w   = (const float*)d_in[10];
    const float* gru_h_b   = (const float*)d_in[11];
    const float* norm2_w   = (const float*)d_in[12];
    const float* fc1_w     = (const float*)d_in[13];
    const float* fc1_b     = (const float*)d_in[14];
    const float* fc2_w     = (const float*)d_in[15];
    const float* fc2_b     = (const float*)d_in[16];
    const float* norm_w    = (const float*)d_in[17];
    const float* out_w     = (const float*)d_in[18];
    const float* out_b     = (const float*)d_in[19];
    float* logits = (float*)d_out;

    constexpr int Vv = 32000, D = 1024, H = 1024, Ll = 6, Bb = 2, Ss = 2048;
    const int M = Bb * Ss;   // 4096

    char* ws = (char*)d_ws; size_t off = 0;
    auto alloc = [&](size_t bytes) -> void* {
        void* p = ws + off; off = (off + bytes + 255) & ~(size_t)255; return p;
    };
    float*  x     = (float*)alloc((size_t)M * D * 4);
    __bf16* ybf   = (__bf16*)alloc((size_t)M * D * 2);
    __bf16* xnbf  = (__bf16*)alloc((size_t)M * D * 2);
    __bf16* h1bf  = (__bf16*)alloc((size_t)M * H * 2);
    float*  kbuf  = (float*)alloc((size_t)M * H * 4);
    float*  thbuf = (float*)alloc((size_t)M * H * 4);
    __bf16* pw_bf = (__bf16*)alloc((size_t)Ll * D * D * 2);
    __bf16* zw_bf = (__bf16*)alloc((size_t)Ll * D * H * 2);
    __bf16* hw_bf = (__bf16*)alloc((size_t)Ll * D * H * 2);
    __bf16* f1_bf = (__bf16*)alloc((size_t)Ll * D * H * 2);
    __bf16* f2_bf = (__bf16*)alloc((size_t)Ll * H * D * 2);
    __bf16* ow_bf = (__bf16*)alloc((size_t)D * Vv * 2);

    auto cvt = [&](const float* s, __bf16* d, size_t n) {
        size_t n4 = n / 4;
        cvt_kernel<<<(unsigned)((n4 + 255) / 256), 256, 0, stream>>>(s, d, (int)n4);
    };
    cvt(conv_pw_w, pw_bf, (size_t)Ll * D * D);
    cvt(gru_z_w,   zw_bf, (size_t)Ll * D * H);
    cvt(gru_h_w,   hw_bf, (size_t)Ll * D * H);
    cvt(fc1_w,     f1_bf, (size_t)Ll * D * H);
    cvt(fc2_w,     f2_bf, (size_t)Ll * H * D);
    cvt(out_w,     ow_bf, (size_t)D * Vv);

    embed_kernel<<<(unsigned)((size_t)M * D / 4 / 256), 256, 0, stream>>>(tokens, emb, x);

    dim3 g1(D / BN, M / BM);
    for (int l = 0; l < Ll; ++l) {
        dwconv_kernel<<<(unsigned)((size_t)M * D / 256), 256, 0, stream>>>(
            x, conv_dw_w + (size_t)l * 3 * D, conv_dw_b + (size_t)l * D, ybf);
        gemm_bf16<1><<<g1, 256, 0, stream>>>(ybf, pw_bf + (size_t)l * D * D,
            conv_pw_b + (size_t)l * D, x, nullptr, M, D, D);
        rmsnorm_kernel<<<M, 256, 0, stream>>>(x, norm1_w + (size_t)l * D, xnbf);
        gemm_bf16<0><<<g1, 256, 0, stream>>>(xnbf, zw_bf + (size_t)l * D * H,
            gru_z_b + (size_t)l * H, kbuf, nullptr, M, H, D);
        gemm_bf16<0><<<g1, 256, 0, stream>>>(xnbf, hw_bf + (size_t)l * D * H,
            gru_h_b + (size_t)l * H, thbuf, nullptr, M, H, D);
        scan_kernel<<<(Bb * H) / 256, 256, 0, stream>>>(kbuf, thbuf, h_prev, x);
        rmsnorm_kernel<<<M, 256, 0, stream>>>(x, norm2_w + (size_t)l * D, xnbf);
        gemm_bf16<2><<<g1, 256, 0, stream>>>(xnbf, f1_bf + (size_t)l * D * H,
            fc1_b + (size_t)l * H, nullptr, h1bf, M, H, D);
        gemm_bf16<1><<<g1, 256, 0, stream>>>(h1bf, f2_bf + (size_t)l * H * D,
            fc2_b + (size_t)l * D, x, nullptr, M, D, H);
    }
    rmsnorm_kernel<<<M, 256, 0, stream>>>(x, norm_w, xnbf);
    dim3 g2(Vv / BN, M / BM);
    gemm_bf16<3><<<g2, 256, 0, stream>>>(xnbf, ow_bf, out_b, logits, nullptr, M, Vv, D);
}
